// ODEVAE_60919816126819
// MI455X (gfx1250) — matrix operand order repaired and checked by reading, not compiled, not run on hardware
//
#include <hip/hip_runtime.h>
#include <hip/hip_bf16.h>

typedef __attribute__((ext_vector_type(16))) _Float16 v16h;
typedef __attribute__((ext_vector_type(8)))  float    v8f;
typedef __attribute__((ext_vector_type(8)))  _Float16 h8;
typedef __attribute__((ext_vector_type(4)))  _Float16 h4;

// ---------------- output layout (f32, concatenated in return order) ----------
#define XREC_N  4014080              // 128*40*1*28*28
#define QM_OFF  4014080              // 128*8
#define QV_OFF  4015104              // 128*8
#define ZT_OFF  4016128              // 128*40*8
#define LH_IDX  4057088
#define KL_IDX  4057089

#define NT_ALL  5120                 // N*T
#define EPSL    1e-5f

// =====================================================================
//  init / finalize
// =====================================================================
__global__ void init_misc(float* out, _Float16* zpad) {
    int t = threadIdx.x;
    if (t < 128) zpad[t] = (_Float16)0.f;    // zero-pad row for OOB gathers
    if (t == 0) { out[LH_IDX] = 0.f; out[KL_IDX] = 0.f; }
}

__global__ void finalize_k(float* out, const int* Ndata) {
    float nd = (float)(*Ndata);
    out[LH_IDX] *= nd / 128.f;     // Ndata * (sum_all / N), N = 128
    out[KL_IDX] *= nd / 1024.f;    // Ndata * mean over 128*8
}

// =====================================================================
//  encoder: direct convs (tiny), NHWC raw f32 -> BN -> f16 NHWC
// =====================================================================
__global__ void enc_conv1(const float* __restrict__ X, const float* __restrict__ w,
                          float* __restrict__ raw) {
    int idx = blockIdx.x * blockDim.x + threadIdx.x;     // 128*14*14*16
    int oc = idx & 15; int r = idx >> 4;
    int ox = r % 14; r /= 14; int oy = r % 14; int n = r / 14;
    float a = 0.f;
    for (int ky = 0; ky < 5; ++ky) {
        int iy = oy * 2 + ky - 2; if (iy < 0 || iy >= 28) continue;
        for (int kx = 0; kx < 5; ++kx) {
            int ix = ox * 2 + kx - 2; if (ix < 0 || ix >= 28) continue;
            a += X[n * 40 * 784 + iy * 28 + ix] * w[oc * 25 + ky * 5 + kx];
        }
    }
    raw[idx] = a;   // bias cancels under batch-norm
}

__global__ void enc_conv2(const _Float16* __restrict__ act, const float* __restrict__ w,
                          float* __restrict__ raw) {
    int idx = blockIdx.x * blockDim.x + threadIdx.x;     // 128*7*7*32
    int oc = idx & 31; int r = idx >> 5;
    int ox = r % 7; r /= 7; int oy = r % 7; int n = r / 7;
    float a = 0.f;
    for (int ky = 0; ky < 5; ++ky) {
        int iy = oy * 2 + ky - 2; if (iy < 0 || iy >= 14) continue;
        for (int kx = 0; kx < 5; ++kx) {
            int ix = ox * 2 + kx - 2; if (ix < 0 || ix >= 14) continue;
            const _Float16* p = act + ((n * 14 + iy) * 14 + ix) * 16;
            const float* wp = w + oc * 16 * 25 + ky * 5 + kx;
            for (int ic = 0; ic < 16; ++ic) a += (float)p[ic] * wp[ic * 25];
        }
    }
    raw[idx] = a;
}

__global__ void enc_conv3(const _Float16* __restrict__ act, const float* __restrict__ w,
                          const float* __restrict__ b, float* __restrict__ henc) {
    int idx = blockIdx.x * blockDim.x + threadIdx.x;     // 128*4*4*64
    int oc = idx & 63; int r = idx >> 6;
    int ox = r & 3; int oy = (r >> 2) & 3; int n = r >> 4;
    float a = b[oc];
    for (int ky = 0; ky < 5; ++ky) {
        int iy = oy * 2 + ky - 2; if (iy < 0 || iy >= 7) continue;
        for (int kx = 0; kx < 5; ++kx) {
            int ix = ox * 2 + kx - 2; if (ix < 0 || ix >= 7) continue;
            const _Float16* p = act + ((n * 7 + iy) * 7 + ix) * 32;
            const float* wp = w + oc * 32 * 25 + ky * 5 + kx;
            for (int ic = 0; ic < 32; ++ic) a += (float)p[ic] * wp[ic * 25];
        }
    }
    henc[idx] = fmaxf(a, 0.f);
}

// =====================================================================
//  batch-norm (training mode, batch stats, biased variance)
// =====================================================================
template<typename T>
__global__ void bn_stats(const T* __restrict__ x, float* __restrict__ stats,
                         int M, int C) {
    int c = blockIdx.x;
    __shared__ float s1[256], s2[256];
    float a = 0.f, b = 0.f;
    for (long r = threadIdx.x; r < M; r += 256) {
        float v = (float)x[r * (long)C + c];
        a += v; b += v * v;
    }
    s1[threadIdx.x] = a; s2[threadIdx.x] = b; __syncthreads();
    for (int s = 128; s > 0; s >>= 1) {
        if (threadIdx.x < s) { s1[threadIdx.x] += s1[threadIdx.x + s];
                               s2[threadIdx.x] += s2[threadIdx.x + s]; }
        __syncthreads();
    }
    if (threadIdx.x == 0) {
        float mean = s1[0] / (float)M;
        float var  = s2[0] / (float)M - mean * mean;
        stats[2 * c]     = mean;
        stats[2 * c + 1] = rsqrtf(var + 1e-5f);
    }
}

template<typename T>
__global__ void bn_apply(const T* __restrict__ x, _Float16* __restrict__ y,
                         const float* __restrict__ stats, const float* __restrict__ g,
                         const float* __restrict__ be, long total, int C) {
    long i = (long)blockIdx.x * blockDim.x + threadIdx.x;
    if (i >= total) return;
    int c = (int)(i % C);
    float v = (float)x[i];
    v = g[c] * (v - stats[2 * c]) * stats[2 * c + 1] + be[c];
    y[i] = (_Float16)fmaxf(v, 0.f);
}

// =====================================================================
//  FC heads + reparameterization + KL
// =====================================================================
__global__ void fc_enc(const float* __restrict__ henc,
                       const float* __restrict__ fc1w, const float* __restrict__ fc1b,
                       const float* __restrict__ fc2w, const float* __restrict__ fc2b,
                       float* __restrict__ qm, float* __restrict__ qv) {
    int idx = blockIdx.x * blockDim.x + threadIdx.x;     // 128*16
    int n = idx >> 4; int j = idx & 15; int jj = j & 7;
    const float* wrow = (j < 8 ? fc1w : fc2w) + jj * 1024;
    float a = (j < 8 ? fc1b : fc2b)[jj];
    for (int i = 0; i < 1024; ++i) {
        int c = i >> 4, y = (i >> 2) & 3, x = i & 3;       // h = NCHW flatten
        a += henc[((n * 4 + y) * 4 + x) * 64 + c] * wrow[i];
    }
    if (j < 8) qm[n * 8 + jj] = a;
    else       qv[n * 8 + jj] = fmaxf(a, 0.f) + log1pf(expf(-fabsf(a)));  // softplus
}

__global__ void z0_kl(const float* __restrict__ qm, const float* __restrict__ qv,
                      const float* __restrict__ eps, float* __restrict__ z0,
                      float* __restrict__ klslot) {
    __shared__ float red[256];
    int i = blockIdx.x * 256 + threadIdx.x;              // 1024 total
    float m = qm[i], v = qv[i];
    z0[i] = m + eps[i] * v;
    red[threadIdx.x] = -logf(v) + 0.5f * (v * v + m * m - 1.f);
    __syncthreads();
    for (int s = 128; s > 0; s >>= 1) {
        if (threadIdx.x < s) red[threadIdx.x] += red[threadIdx.x + s];
        __syncthreads();
    }
    if (threadIdx.x == 0) atomicAdd(klslot, red[0]);
}

// =====================================================================
//  RK4 scan of the tiny MLP vector field (one persistent block)
// =====================================================================
__device__ void mlp_eval(const float* z, float* o,
                         const float* sW1, const float* sB1,
                         const float* sW2, const float* sB2,
                         const float* sW3, const float* sB3) {
    float h1[100], h2[100];
    for (int i = 0; i < 100; ++i) {
        float a = sB1[i];
        for (int q = 0; q < 8; ++q) a += z[q] * sW1[i * 8 + q];
        h1[i] = fmaxf(a, 0.f);
    }
    for (int i = 0; i < 100; ++i) {
        float a = sB2[i];
        for (int j = 0; j < 100; ++j) a += h1[j] * sW2[i * 100 + j];
        h2[i] = fmaxf(a, 0.f);
    }
    for (int q = 0; q < 8; ++q) {
        float a = sB3[q];
        for (int j = 0; j < 100; ++j) a += h2[j] * sW3[q * 100 + j];
        o[q] = a;
    }
}

__global__ void ode_rk4(const float* __restrict__ z0,
                        const float* __restrict__ w1, const float* __restrict__ b1,
                        const float* __restrict__ w2, const float* __restrict__ b2,
                        const float* __restrict__ w3, const float* __restrict__ b3,
                        float* __restrict__ zt) {
    __shared__ float sW1[800], sB1[100], sW2[10000], sB2[100], sW3[800], sB3[8];
    for (int i = threadIdx.x; i < 800;   i += blockDim.x) { sW1[i] = w1[i]; sW3[i] = w3[i]; }
    for (int i = threadIdx.x; i < 100;   i += blockDim.x) { sB1[i] = b1[i]; sB2[i] = b2[i]; }
    for (int i = threadIdx.x; i < 10000; i += blockDim.x) sW2[i] = w2[i];
    if (threadIdx.x < 8) sB3[threadIdx.x] = b3[threadIdx.x];
    __syncthreads();

    int n = threadIdx.x;                                 // 128 threads = 128 samples
    float z[8];
    for (int q = 0; q < 8; ++q) { z[q] = z0[n * 8 + q]; zt[(n * 40) * 8 + q] = z[q]; }
    const float dt = 0.1f;
    for (int s = 0; s < 39; ++s) {
        float k1[8], k2[8], k3[8], k4[8], tz[8];
        mlp_eval(z, k1, sW1, sB1, sW2, sB2, sW3, sB3);
        for (int q = 0; q < 8; ++q) tz[q] = z[q] + 0.5f * dt * k1[q];
        mlp_eval(tz, k2, sW1, sB1, sW2, sB2, sW3, sB3);
        for (int q = 0; q < 8; ++q) tz[q] = z[q] + 0.5f * dt * k2[q];
        mlp_eval(tz, k3, sW1, sB1, sW2, sB2, sW3, sB3);
        for (int q = 0; q < 8; ++q) tz[q] = z[q] + dt * k3[q];
        mlp_eval(tz, k4, sW1, sB1, sW2, sB2, sW3, sB3);
        for (int q = 0; q < 8; ++q) {
            z[q] += (dt / 6.f) * (k1[q] + 2.f * k2[q] + 2.f * k3[q] + k4[q]);
            zt[(n * 40 + s + 1) * 8 + q] = z[q];
        }
    }
}

// =====================================================================
//  fc3: [NT,8] @ [8,1024] -> pack to NHWC f16 [NT,4,4,64]
// =====================================================================
__global__ void fc3_pack(const float* __restrict__ zt, const float* __restrict__ w,
                         const float* __restrict__ b, _Float16* __restrict__ g0) {
    int idx = blockIdx.x * blockDim.x + threadIdx.x;     // 5120*1024
    int j = idx & 1023; int nt = idx >> 10;
    const float* zr = zt + nt * 8;
    const float* wr = w + j * 8;
    float a = b[j];
    for (int q = 0; q < 8; ++q) a += zr[q] * wr[q];
    int c = j >> 4, y = (j >> 2) & 3, x = j & 3;          // reshape (NT,64,4,4)
    g0[((nt * 4 + y) * 4 + x) * 64 + c] = (_Float16)a;
}

// =====================================================================
//  decoder weight prep -> fragment-ready B layout:
//  wF[(((t*(IC/32)+ch)*(OC/16))+octile)*512 + n*32 + k]
//     = flipped convT weight for (ic = ch*32+k, oc = octile*16+n, tap t)
//  Each lane's B fragment is then two 16-byte runs, 32 bytes apart, in global memory.
// =====================================================================
__global__ void prep_wt_frag(const float* __restrict__ w, _Float16* __restrict__ wF,
                             int IC, int OC, int KH, int KW) {
    int idx = blockIdx.x * blockDim.x + threadIdx.x;     // KH*KW*IC*OC
    int e = idx & 511; int r = idx >> 9;
    int n = e >> 5; int k = e & 31;
    int noct = OC >> 4;
    int octile = r % noct; int r2 = r / noct;
    int nch = IC >> 5;
    int ch = r2 % nch; int t = r2 / nch;
    int ky = t / KW, kx = t % KW;
    int ic = ch * 32 + k; int oc = octile * 16 + n;
    wF[idx] = (_Float16)w[((ic * OC + oc) * KH + (KH - 1 - ky)) * KW + (KW - 1 - kx)];
}

// =====================================================================
//  Implicit-GEMM transposed conv with WMMA (f16 in, f32 acc, f16 out)
//  - act NHWC f16 [NT,IH,IW,IC]; out raw f16 [NT*OH*OW, OC]
//  - one wave = one 16x16 D tile; waves split OC; K = taps x IC chunks of 32
//  - A staged once per tap into LDS (full 16 x IC row set); fragments read
//    as 2x ds_load_b128 per lane (contiguous per CDNA5 A layout)
//  - B fragments loaded directly from global (fragment-ready layout)
//  - OOB / dilation-gap rows redirect to a zero page (branchless select)
// =====================================================================
template<int IC, int KH, int KW, int S, int LO>
__global__ void convt_gemm_t(const _Float16* __restrict__ act,
                             const _Float16* __restrict__ wF,
                             const _Float16* __restrict__ zpad,
                             _Float16* __restrict__ outRaw,
                             int IH, int IW, int OH, int OW, int OC) {
    constexpr int NCH = IC / 32;
    __shared__ _Float16 AT[16][IC + 8];                  // padded row stride
    __shared__ const _Float16* rowSrc[16];
    __shared__ int rNT[16], rOY[16], rOX[16];

    const int tid    = threadIdx.x;
    const int lane   = tid & 31;
    const int w      = tid >> 5;
    const int nW     = blockDim.x >> 5;
    const int octile = blockIdx.y * nW + w;
    const int oc0    = octile * 16;
    const int m0     = blockIdx.x * 16;
    const int HW     = OH * OW;
    const int noct   = OC >> 4;

    if (tid < 16) {
        int mg = m0 + tid;
        int nt = mg / HW; int r = mg - nt * HW;
        rNT[tid] = nt; rOY[tid] = r / OW; rOX[tid] = r - (r / OW) * OW;
    }

    v8f acc = {};
    const int mm   = lane & 15;
    const int half = lane >> 4;

    for (int t = 0; t < KH * KW; ++t) {
        const int ky = t / KW, kx = t - (t / KW) * KW;
        __syncthreads();                                 // prior frag reads done
        if (tid < 16) {
            int yd = rOY[tid] + ky - LO;
            int xd = rOX[tid] + kx - LO;
            bool ok = (yd >= 0) && (xd >= 0);
            int iy, ix;
            if (S == 2) { ok = ok && !(yd & 1) && !(xd & 1); iy = yd >> 1; ix = xd >> 1; }
            else        { iy = yd; ix = xd; }
            ok = ok && (iy < IH) && (ix < IW);
            rowSrc[tid] = ok ? act + ((size_t)(rNT[tid] * IH + iy) * IW + ix) * IC
                             : zpad;
        }
        __syncthreads();
        // ---- stage the full 16 x IC A-row set for this tap
        for (int idx = tid; idx < 16 * (IC / 4); idx += blockDim.x) {
            int m  = idx / (IC / 4);
            int jj = (idx - m * (IC / 4)) * 4;
            h4 v = *(const h4*)(rowSrc[m] + jj);
            *(h4*)(&AT[m][jj]) = v;
        }
        // prefetch next tap's weight fragments (L2 temporal)
        if (t + 1 < KH * KW)
            __builtin_prefetch(wF + ((size_t)((t + 1) * NCH) * noct + octile) * 512
                                   + mm * 32, 0, 1);
        __syncthreads();
        // ---- unrolled K-chunk WMMAs for this tap
        #pragma unroll
        for (int ch = 0; ch < NCH; ++ch) {
            h8 a0 = *(const h8*)(&AT[mm][ch * 32 + half * 8]);
            h8 a1 = *(const h8*)(&AT[mm][ch * 32 + 16 + half * 8]);
            const _Float16* bp = wF + ((size_t)(t * NCH + ch) * noct + octile) * 512
                                    + mm * 32 + half * 8;
            h8 b0 = *(const h8*)(bp);
            h8 b1 = *(const h8*)(bp + 16);
            v16h afrag, bfrag;
            #pragma unroll
            for (int i = 0; i < 8; ++i) {
                afrag[i] = a0[i]; afrag[8 + i] = a1[i];
                bfrag[i] = b0[i]; bfrag[8 + i] = b1[i];
            }
            acc = __builtin_amdgcn_wmma_f32_16x16x32_f16(
                      false, afrag, false, bfrag, (short)0, acc, false, false);
        }
    }
    // ---- store D: VGPR r -> (M = r + half*8, N = lane&15)
    #pragma unroll
    for (int r = 0; r < 8; ++r) {
        int m = r + (half << 3);
        size_t row = (size_t)(m0 + m);
        outRaw[row * OC + oc0 + mm] = (_Float16)acc[r];
    }
}

// =====================================================================
//  final conv (32->1, 5x5, p=2) + sigmoid + Bernoulli log-lik reduction
// =====================================================================
__global__ void dec4_loss(const _Float16* __restrict__ act,
                          const float* __restrict__ w4, const float* __restrict__ b4,
                          const float* __restrict__ X, float* __restrict__ out) {
    __shared__ float sw[800];          // flipped weights: sw[(ky*5+kx)*32+ic]
    __shared__ float red[256];
    for (int s = threadIdx.x; s < 800; s += 256) {
        int t = s / 32, ic = s % 32;
        int ky = t / 5, kx = t % 5;
        sw[s] = w4[ic * 25 + (4 - ky) * 5 + (4 - kx)];
    }
    __syncthreads();

    int idx = blockIdx.x * 256 + threadIdx.x;            // 5120*784 exactly
    int nt = idx / 784; int r = idx - nt * 784;
    int oy = r / 28, ox = r - (r / 28) * 28;
    float a = b4[0];
    for (int ky = 0; ky < 5; ++ky) {
        int iy = oy + ky - 2; if (iy < 0 || iy >= 28) continue;
        for (int kx = 0; kx < 5; ++kx) {
            int ix = ox + kx - 2; if (ix < 0 || ix >= 28) continue;
            const _Float16* p = act + ((nt * 28 + iy) * 28 + ix) * 32;
            const float* wp = sw + (ky * 5 + kx) * 32;
            for (int ic = 0; ic < 32; ++ic) a += (float)p[ic] * wp[ic];
        }
    }
    float xr = 1.f / (1.f + expf(-a));
    out[idx] = xr;                                       // Xrec
    float xv = X[idx];
    red[threadIdx.x] = logf(EPSL + xr) * xv + logf(EPSL + 1.f - xr) * (1.f - xv);
    __syncthreads();
    for (int s = 128; s > 0; s >>= 1) {
        if (threadIdx.x < s) red[threadIdx.x] += red[threadIdx.x + s];
        __syncthreads();
    }
    if (threadIdx.x == 0) atomicAdd(out + LH_IDX, red[0]);
}

// =====================================================================
//  host-side orchestration
// =====================================================================
extern "C" void kernel_launch(void* const* d_in, const int* in_sizes, int n_in,
                              void* d_out, int out_size, void* d_ws, size_t ws_size,
                              hipStream_t stream) {
    (void)in_sizes; (void)n_in; (void)out_size; (void)ws_size;
    // ---- inputs (setup_inputs order; params dict flattened in insertion order)
    const float* X      = (const float*)d_in[0];
    const float* eps_z  = (const float*)d_in[1];
    const int*   Ndata  = (const int*)d_in[2];
    const float* enc_w1 = (const float*)d_in[3];
    const float* enc_g1 = (const float*)d_in[5];
    const float* enc_be1= (const float*)d_in[6];
    const float* enc_w2 = (const float*)d_in[7];
    const float* enc_g2 = (const float*)d_in[9];
    const float* enc_be2= (const float*)d_in[10];
    const float* enc_w3 = (const float*)d_in[11];
    const float* enc_b3 = (const float*)d_in[12];
    const float* fc1_w  = (const float*)d_in[13];
    const float* fc1_b  = (const float*)d_in[14];
    const float* fc2_w  = (const float*)d_in[15];
    const float* fc2_b  = (const float*)d_in[16];
    const float* f_w1   = (const float*)d_in[17];
    const float* f_b1   = (const float*)d_in[18];
    const float* f_w2   = (const float*)d_in[19];
    const float* f_b2   = (const float*)d_in[20];
    const float* f_w3   = (const float*)d_in[21];
    const float* f_b3   = (const float*)d_in[22];
    const float* fc3_w  = (const float*)d_in[23];
    const float* fc3_b  = (const float*)d_in[24];
    const float* dec_w1 = (const float*)d_in[25];
    const float* dec_g1 = (const float*)d_in[27];
    const float* dec_be1= (const float*)d_in[28];
    const float* dec_w2 = (const float*)d_in[29];
    const float* dec_g2 = (const float*)d_in[31];
    const float* dec_be2= (const float*)d_in[32];
    const float* dec_w3 = (const float*)d_in[33];
    const float* dec_g3 = (const float*)d_in[35];
    const float* dec_be3= (const float*)d_in[36];
    const float* dec_w4 = (const float*)d_in[37];
    const float* dec_b4 = (const float*)d_in[38];

    float* out = (float*)d_out;

    // ---- workspace layout (ping-pong A/B + small buffers)
    char* ws = (char*)d_ws;
    const size_t A_SZ  = (size_t)NT_ALL * 784 * 32 * 2;  // 256,901,120 (max act)
    size_t off = 0;
    _Float16* actA  = (_Float16*)(ws + off); off += A_SZ;  // G0c/H1c/H2c/H3c
    _Float16* rawB  = (_Float16*)(ws + off); off += A_SZ;  // R1/R2/R3 (f16 raw)
    _Float16* wF1   = (_Float16*)(ws + off); off += 576  * 128 * 2;
    _Float16* wF2   = (_Float16*)(ws + off); off += 3200 * 64  * 2;
    _Float16* wF3   = (_Float16*)(ws + off); off += 1600 * 32  * 2;
    float*    stats = (float*)   (ws + off); off += 256 * 4;
    _Float16* zpad  = (_Float16*)(ws + off); off += 128 * 2;
    float*    eraw1 = (float*)   (ws + off); off += (size_t)128 * 196 * 16 * 4;
    _Float16* e1    = (_Float16*)(ws + off); off += (size_t)128 * 196 * 16 * 2;
    float*    eraw2 = (float*)   (ws + off); off += (size_t)128 * 49 * 32 * 4;
    _Float16* e2    = (_Float16*)(ws + off); off += (size_t)128 * 49 * 32 * 2;
    float*    henc  = (float*)   (ws + off); off += (size_t)128 * 16 * 64 * 4;
    float*    z0buf = (float*)   (ws + off); off += 1024 * 4;

    init_misc<<<1, 128, 0, stream>>>(out, zpad);

    // ---- encoder (first frame) ----
    enc_conv1<<<1568, 256, 0, stream>>>(X, enc_w1, eraw1);
    bn_stats<float><<<16, 256, 0, stream>>>(eraw1, stats, 128 * 196, 16);
    bn_apply<float><<<1568, 256, 0, stream>>>(eraw1, e1, stats, enc_g1, enc_be1,
                                              (long)401408, 16);
    enc_conv2<<<784, 256, 0, stream>>>(e1, enc_w2, eraw2);
    bn_stats<float><<<32, 256, 0, stream>>>(eraw2, stats, 128 * 49, 32);
    bn_apply<float><<<784, 256, 0, stream>>>(eraw2, e2, stats, enc_g2, enc_be2,
                                             (long)200704, 32);
    enc_conv3<<<512, 256, 0, stream>>>(e2, enc_w3, enc_b3, henc);
    fc_enc<<<8, 256, 0, stream>>>(henc, fc1_w, fc1_b, fc2_w, fc2_b,
                                  out + QM_OFF, out + QV_OFF);
    z0_kl<<<4, 256, 0, stream>>>(out + QM_OFF, out + QV_OFF, eps_z, z0buf,
                                 out + KL_IDX);

    // ---- RK4 ODE scan (sequential, one persistent block) ----
    ode_rk4<<<1, 128, 0, stream>>>(z0buf, f_w1, f_b1, f_w2, f_b2, f_w3, f_b3,
                                   out + ZT_OFF);

    // ---- decoder weight prep (flip + transpose to fragment-ready f16) ----
    prep_wt_frag<<<288, 256, 0, stream>>>(dec_w1, wF1, 64, 128, 3, 3);
    prep_wt_frag<<<800, 256, 0, stream>>>(dec_w2, wF2, 128, 64, 5, 5);
    prep_wt_frag<<<200, 256, 0, stream>>>(dec_w3, wF3, 64, 32, 5, 5);

    // ---- fc3 + pack to NHWC f16 [NT,4,4,64] ----
    fc3_pack<<<20480, 256, 0, stream>>>(out + ZT_OFF, fc3_w, fc3_b, actA);

    // ---- decoder layer 1: 64 -> 128, k3 s1 p0 (4x4 -> 6x6) ----
    convt_gemm_t<64, 3, 3, 1, 2><<<dim3(11520, 2), 128, 0, stream>>>(
        actA, wF1, zpad, rawB, 4, 4, 6, 6, 128);
    bn_stats<_Float16><<<128, 256, 0, stream>>>(rawB, stats, NT_ALL * 36, 128);
    bn_apply<_Float16><<<92160, 256, 0, stream>>>(rawB, actA, stats, dec_g1, dec_be1,
                                                  (long)NT_ALL * 36 * 128, 128);

    // ---- decoder layer 2: 128 -> 64, k5 s2 p1 (6x6 -> 13x13) ----
    convt_gemm_t<128, 5, 5, 2, 3><<<dim3(54080, 1), 128, 0, stream>>>(
        actA, wF2, zpad, rawB, 6, 6, 13, 13, 64);
    bn_stats<_Float16><<<64, 256, 0, stream>>>(rawB, stats, NT_ALL * 169, 64);
    bn_apply<_Float16><<<216320, 256, 0, stream>>>(rawB, actA, stats, dec_g2, dec_be2,
                                                   (long)NT_ALL * 169 * 64, 64);

    // ---- decoder layer 3: 64 -> 32, k5 s2 p1 op1 (13x13 -> 28x28) ----
    convt_gemm_t<64, 5, 5, 2, 3><<<dim3(250880, 1), 64, 0, stream>>>(
        actA, wF3, zpad, rawB, 13, 13, 28, 28, 32);
    bn_stats<_Float16><<<32, 256, 0, stream>>>(rawB, stats, NT_ALL * 784, 32);
    bn_apply<_Float16><<<501760, 256, 0, stream>>>(rawB, actA, stats, dec_g3, dec_be3,
                                                   (long)NT_ALL * 784 * 32, 32);

    // ---- final conv + sigmoid + Bernoulli log-lik ----
    dec4_loss<<<15680, 256, 0, stream>>>(actA, dec_w4, dec_b4, X, out);

    finalize_k<<<1, 1, 0, stream>>>(out, Ndata);
}
